// Attention_10625749090555
// MI455X (gfx1250) — compile-verified, matrix operand order checked by reading, not run on hardware
//
#include <hip/hip_runtime.h>

#define B_SZ 2
#define L_SEQ 2048
#define D_MODEL 2048
#define NH 32
#define KVH 8
#define HD 64

typedef __attribute__((ext_vector_type(16))) _Float16 v16h;
typedef __attribute__((ext_vector_type(2)))  _Float16 h2;
typedef __attribute__((ext_vector_type(8)))  float    v8f;

// k-offset of half-pair j within a 16x32 f16 A/B fragment, per ISA 7.12.2:
// VGPRs 0-3 hold K = g*8 + {0,2,4,6}+{0,1}; VGPRs 4-7 hold K = 16 + g*8 + ...
__device__ __forceinline__ int kbase(int j, int g) {
  return ((j >> 2) << 4) + (g << 3) + ((j & 3) << 1);
}

// Build a 16x32 f16 fragment from a row-major f16 matrix (global or LDS).
// `row` is this lane's row (A: output row m = lane%16; B: output column n).
__device__ __forceinline__ v16h frag_f16(const _Float16* __restrict__ base, int ld,
                                         int row, int k0, int g) {
  v16h f;
  const _Float16* p = base + (size_t)row * ld + k0;
#pragma unroll
  for (int j = 0; j < 8; ++j) {
    const int k = kbase(j, g);
    const h2 t = *(const h2*)(p + k);
    f[2 * j]     = t[0];
    f[2 * j + 1] = t[1];
  }
  return f;
}

__device__ __forceinline__ v8f wmma16(v16h a, v16h b, v8f c) {
  return __builtin_amdgcn_wmma_f32_16x16x32_f16(false, a, false, b, (short)0, c,
                                                false, false);
}

// Async DMA of 16 bytes/lane into LDS (gfx1250 ASYNCcnt path). Low 32 bits of
// a flat LDS pointer are the DS address (ISA 10.2 aperture mapping), so the
// truncated generic pointer is the correct VDST value.
__device__ __forceinline__ void async_b128(const void* gptr, void* lptr) {
  asm volatile("global_load_async_to_lds_b128 %0, %1, off"
               :: "v"((unsigned)(unsigned long long)(size_t)lptr),
                  "v"((unsigned long long)(size_t)gptr)
               : "memory");
}

// ---------------------------------------------------------------------------
// Kernel 0: one-shot f32 -> f16 conversion (x and all weights). Removes all
// v_cvt_pk_f16_f32 from the GEMM hot loops and halves their load bytes.
// ---------------------------------------------------------------------------
__global__ __launch_bounds__(256) void cvt_f16_kernel(
    const float* __restrict__ src, _Float16* __restrict__ dst, int n4) {
  const int i = blockIdx.x * 256 + threadIdx.x;
  if (i < n4) {
    const float4 v = ((const float4*)src)[i];
    h2 a, b;
    a[0] = (_Float16)v.x; a[1] = (_Float16)v.y;
    b[0] = (_Float16)v.z; b[1] = (_Float16)v.w;
    ((h2*)dst)[2 * i]     = a;
    ((h2*)dst)[2 * i + 1] = b;
  }
}

// ---------------------------------------------------------------------------
// Kernel 1: fused QKV projection + RoPE (Q,K) + transpose-store (V).
// One wave computes a 16-row x 64-col strip = one full head of one row-tile.
// grid = (B*L/16, 32 Q-strips + 8 K-strips + 8 V-strips)
// ---------------------------------------------------------------------------
__global__ __launch_bounds__(32) void qkv_rope_kernel(
    const _Float16* __restrict__ x, const _Float16* __restrict__ wq,
    const _Float16* __restrict__ wk, const _Float16* __restrict__ wv,
    _Float16* __restrict__ Q, _Float16* __restrict__ K,
    _Float16* __restrict__ Vt) {
  const int lane = threadIdx.x;
  const int g = lane >> 4, ln = lane & 15;
  const int row0 = blockIdx.x * 16;  // flat (b*L + l) tile
  const int strip = blockIdx.y;

  const _Float16* W;
  int mode, head;
  if (strip < 32)      { W = wq; mode = 0; head = strip; }
  else if (strip < 40) { W = wk; mode = 1; head = strip - 32; }
  else                 { W = wv; mode = 2; head = strip - 40; }
  const int e0 = head * HD;

  v8f acc[4] = {{}, {}, {}, {}};
  for (int d0 = 0; d0 < D_MODEL; d0 += 32) {
    __builtin_prefetch(x + (size_t)(row0 + ln) * D_MODEL + d0 + 256, 0, 1);
    __builtin_prefetch(W + (size_t)(e0 + ln) * D_MODEL + d0 + 256, 0, 1);
    const v16h a = frag_f16(x, D_MODEL, row0 + ln, d0, g);
#pragma unroll
    for (int t = 0; t < 4; ++t) {
      const v16h b = frag_f16(W, D_MODEL, e0 + t * 16 + ln, d0, g);
      acc[t] = wmma16(a, b, acc[t]);
    }
  }

  if (mode < 2) {
    // RoPE in-register: this lane holds head-columns d = ln, ln+16, ln+32,
    // ln+48 (fragments 0..3); rotate pairs are (0<->2) and (1<->3), and the
    // cos/sin tables repeat with period 32, so everything is lane-local.
    const float if0 = __powf(10000.0f, -(float)ln / 32.0f);
    const float if1 = __powf(10000.0f, -(float)(ln + 16) / 32.0f);
    _Float16* out = (mode == 0) ? Q : K;
    const int nh = (mode == 0) ? NH : KVH;
    const float qscale = (mode == 0) ? 0.125f : 1.0f;  // fold 1/sqrt(HD) into Q
#pragma unroll
    for (int r = 0; r < 8; ++r) {
      const int row = row0 + r + 8 * g;
      const int b = row >> 11, l = row & (L_SEQ - 1);
      float s0, c0, s1, c1;
      __sincosf((float)l * if0, &s0, &c0);
      __sincosf((float)l * if1, &s1, &c1);
      const float a0 = acc[0][r], a1 = acc[1][r], a2 = acc[2][r], a3 = acc[3][r];
      const float n0 = (c0 * a0 - s0 * a2) * qscale;
      const float n1 = (c1 * a1 - s1 * a3) * qscale;
      const float n2 = (c0 * a2 + s0 * a0) * qscale;
      const float n3 = (c1 * a3 + s1 * a1) * qscale;
      _Float16* p = out + (((size_t)b * nh + head) * L_SEQ + l) * HD;
      p[ln]      = (_Float16)n0;
      p[ln + 16] = (_Float16)n1;
      p[ln + 32] = (_Float16)n2;
      p[ln + 48] = (_Float16)n3;
    }
  } else {
    // V stored transposed [B][KVH][HD][L]: attention V chunks become 64B rows.
#pragma unroll
    for (int t = 0; t < 4; ++t) {
      const int d = t * 16 + ln;
#pragma unroll
      for (int r = 0; r < 8; ++r) {
        const int row = row0 + r + 8 * g;
        const int b = row >> 11, l = row & (L_SEQ - 1);
        Vt[(((size_t)b * KVH + head) * HD + d) * L_SEQ + l] = (_Float16)acc[t][r];
      }
    }
  }
}

// Stage one 32-key chunk of K ([32][64] f16, contiguous 4KB) and V (transposed
// [64][32] f16, 64B rows) into LDS: 16 async b128 ops (16B/lane each).
__device__ __forceinline__ void stage_kv_chunk(const _Float16* Kp,
                                               const _Float16* Vp, int k0,
                                               _Float16* kb, _Float16* vb,
                                               int lane) {
  const char* gk = (const char*)(Kp + (size_t)k0 * HD);
  char* lk = (char*)kb;
#pragma unroll
  for (int i = 0; i < 8; ++i) {
    const int off = i * 512 + lane * 16;
    async_b128(gk + off, lk + off);
  }
#pragma unroll
  for (int i = 0; i < 8; ++i) {
    const int t = i * 32 + lane;         // 0..255 quarter-rows
    const int d = t >> 2, q = (t & 3) * 16;
    async_b128((const char*)(Vp + (size_t)d * L_SEQ + k0) + q,
               (char*)vb + t * 16);
  }
}

// ---------------------------------------------------------------------------
// Kernel 2: streaming causal GQA attention, one wave per (b, head, 16 queries).
// K/V chunks double-buffered in LDS via global_load_async_to_lds_b128 +
// s_wait_asynccnt; 4 WMMA for S = Q*K^T, online softmax with shfl_xor row
// reductions, P transposed C->A layout through LDS, 4 WMMA for P*V.
// grid = (L/16, NH, B)
// ---------------------------------------------------------------------------
__global__ __launch_bounds__(32) void attn_kernel(
    const _Float16* __restrict__ Q, const _Float16* __restrict__ K,
    const _Float16* __restrict__ Vt, _Float16* __restrict__ O) {
  __shared__ _Float16 kbuf[2][32 * HD];
  __shared__ _Float16 vbuf[2][HD * 32];
  __shared__ _Float16 pbuf[16 * 32];
  const int lane = threadIdx.x;
  const int g = lane >> 4, ln = lane & 15;
  const int q0 = blockIdx.x * 16;
  const int h = blockIdx.y;
  const int b = blockIdx.z;
  const int kvh = h >> 2;  // rep = NH/KVH = 4

  const _Float16* Qp = Q + ((size_t)b * NH + h) * L_SEQ * HD;
  const _Float16* Kp = K + ((size_t)b * KVH + kvh) * L_SEQ * HD;
  const _Float16* Vp = Vt + ((size_t)b * KVH + kvh) * HD * L_SEQ;

  const v16h aQ0 = frag_f16(Qp, HD, q0 + ln, 0, g);
  const v16h aQ1 = frag_f16(Qp, HD, q0 + ln, 32, g);

  v8f acc[4] = {{}, {}, {}, {}};
  float mrow[8], lrow[8];
#pragma unroll
  for (int r = 0; r < 8; ++r) { mrow[r] = -3.0e30f; lrow[r] = 0.0f; }

  const int kend = q0 + 16;
  stage_kv_chunk(Kp, Vp, 0, kbuf[0], vbuf[0], lane);

  for (int k0 = 0; k0 < kend; k0 += 32) {
    const int cur = (k0 >> 5) & 1;
    if (k0 + 32 < kend) {
      stage_kv_chunk(Kp, Vp, k0 + 32, kbuf[cur ^ 1], vbuf[cur ^ 1], lane);
      asm volatile("s_wait_asynccnt 0x10" ::: "memory");  // prev 16 done
    } else {
      asm volatile("s_wait_asynccnt 0x0" ::: "memory");
    }
    const _Float16* kb = kbuf[cur];
    const _Float16* vb = vbuf[cur];

    // S (16 queries x 32 keys): B-frag column = key, k = d, from LDS K rows.
    v8f s0 = {}, s1 = {};
    {
      v16h bk = frag_f16(kb, HD, ln, 0, g);
      s0 = wmma16(aQ0, bk, s0);
      bk = frag_f16(kb, HD, ln, 32, g);
      s0 = wmma16(aQ1, bk, s0);
      bk = frag_f16(kb, HD, 16 + ln, 0, g);
      s1 = wmma16(aQ0, bk, s1);
      bk = frag_f16(kb, HD, 16 + ln, 32, g);
      s1 = wmma16(aQ1, bk, s1);
    }
    if (k0 + 32 > q0) {  // only diagonal chunks need the causal mask
#pragma unroll
      for (int r = 0; r < 8; ++r) {
        const int qq = q0 + r + 8 * g;
        if (k0 + ln > qq)      s0[r] = -3.0e30f;
        if (k0 + 16 + ln > qq) s1[r] = -3.0e30f;
      }
    }
    // Online softmax: each row's 16 values live in one 16-lane group at a
    // fixed VGPR index -> xor-reduce with masks 1,2,4,8 (stays in the group).
    float esc[8];
#pragma unroll
    for (int r = 0; r < 8; ++r) {
      float mx = fmaxf(s0[r], s1[r]);
      mx = fmaxf(mx, __shfl_xor(mx, 1, 32));
      mx = fmaxf(mx, __shfl_xor(mx, 2, 32));
      mx = fmaxf(mx, __shfl_xor(mx, 4, 32));
      mx = fmaxf(mx, __shfl_xor(mx, 8, 32));
      const float mn = fmaxf(mrow[r], mx);
      esc[r] = __expf(mrow[r] - mn);
      const float p0 = __expf(s0[r] - mn);
      const float p1 = __expf(s1[r] - mn);
      s0[r] = p0;
      s1[r] = p1;
      float ps = p0 + p1;
      ps += __shfl_xor(ps, 1, 32);
      ps += __shfl_xor(ps, 2, 32);
      ps += __shfl_xor(ps, 4, 32);
      ps += __shfl_xor(ps, 8, 32);
      lrow[r] = lrow[r] * esc[r] + ps;
      mrow[r] = mn;
    }
#pragma unroll
    for (int t = 0; t < 4; ++t)
#pragma unroll
      for (int r = 0; r < 8; ++r) acc[t][r] *= esc[r];

    // Transpose P from C-layout to A-layout through LDS (single wave: DS ops
    // complete in order, compiler inserts s_wait_dscnt).
#pragma unroll
    for (int r = 0; r < 8; ++r) {
      pbuf[(r + 8 * g) * 32 + ln]      = (_Float16)s0[r];
      pbuf[(r + 8 * g) * 32 + 16 + ln] = (_Float16)s1[r];
    }
    v16h aP;
    {
      const _Float16* lp = pbuf + ln * 32;
#pragma unroll
      for (int j = 0; j < 8; ++j) {
        const int k = kbase(j, g);
        aP[2 * j]     = lp[k];
        aP[2 * j + 1] = lp[k + 1];
      }
    }
    // P (16x32) * V-chunk (32 keys x 64 d); B-frag rows = d, from LDS.
#pragma unroll
    for (int t = 0; t < 4; ++t) {
      const v16h bv = frag_f16(vb, 32, t * 16 + ln, 0, g);
      acc[t] = wmma16(aP, bv, acc[t]);
    }
  }

  // Epilogue: O[b][l][h*64 + d] (the [B,L,H*hd] layout the o-proj consumes).
#pragma unroll
  for (int t = 0; t < 4; ++t) {
#pragma unroll
    for (int r = 0; r < 8; ++r) {
      const int l = q0 + r + 8 * g;
      const float o = acc[t][r] / lrow[r];
      O[((size_t)b * L_SEQ + l) * D_MODEL + h * HD + t * 16 + ln] = (_Float16)o;
    }
  }
}

// ---------------------------------------------------------------------------
// Kernel 3: output projection  out = O @ wo^T  (f16 inputs, f32 out).
// grid = (B*L/16, D/64)
// ---------------------------------------------------------------------------
__global__ __launch_bounds__(32) void oproj_kernel(
    const _Float16* __restrict__ O, const _Float16* __restrict__ wo,
    float* __restrict__ out) {
  const int lane = threadIdx.x;
  const int g = lane >> 4, ln = lane & 15;
  const int row0 = blockIdx.x * 16;
  const int e0 = blockIdx.y * 64;

  v8f acc[4] = {{}, {}, {}, {}};
  for (int d0 = 0; d0 < D_MODEL; d0 += 32) {
    __builtin_prefetch(O + (size_t)(row0 + ln) * D_MODEL + d0 + 256, 0, 1);
    __builtin_prefetch(wo + (size_t)(e0 + ln) * D_MODEL + d0 + 256, 0, 1);
    const v16h a = frag_f16(O, D_MODEL, row0 + ln, d0, g);
#pragma unroll
    for (int t = 0; t < 4; ++t) {
      const v16h b = frag_f16(wo, D_MODEL, e0 + t * 16 + ln, d0, g);
      acc[t] = wmma16(a, b, acc[t]);
    }
  }
#pragma unroll
  for (int t = 0; t < 4; ++t)
#pragma unroll
    for (int r = 0; r < 8; ++r)
      out[(size_t)(row0 + r + 8 * g) * D_MODEL + e0 + t * 16 + ln] = acc[t][r];
}

extern "C" void kernel_launch(void* const* d_in, const int* in_sizes, int n_in,
                              void* d_out, int out_size, void* d_ws,
                              size_t ws_size, hipStream_t stream) {
  (void)in_sizes; (void)n_in; (void)out_size; (void)ws_size;
  const float* x  = (const float*)d_in[0];
  const float* wq = (const float*)d_in[1];
  const float* wk = (const float*)d_in[2];
  const float* wv = (const float*)d_in[3];
  const float* wo = (const float*)d_in[4];
  float* out = (float*)d_out;

  // Workspace carve-out (bytes):
  // xh 16M | wqh 8M | wkh 2M | wvh 2M | woh 8M | Q 16M | K 4M | Vt 4M | O 16M
  char* ws = (char*)d_ws;
  const size_t MB = 1024 * 1024;
  _Float16* xh  = (_Float16*)(ws);
  _Float16* wqh = (_Float16*)(ws + 16 * MB);
  _Float16* wkh = (_Float16*)(ws + 24 * MB);
  _Float16* wvh = (_Float16*)(ws + 26 * MB);
  _Float16* woh = (_Float16*)(ws + 28 * MB);
  _Float16* Q   = (_Float16*)(ws + 36 * MB);
  _Float16* K   = (_Float16*)(ws + 52 * MB);
  _Float16* Vt  = (_Float16*)(ws + 56 * MB);
  _Float16* O   = (_Float16*)(ws + 60 * MB);

  const dim3 blk32(32), blk256(256);

  // f32 -> f16 one-shot conversions (bandwidth pass, fits in L2 afterwards).
  const size_t nx = (size_t)B_SZ * L_SEQ * D_MODEL;
  const size_t nqo = (size_t)D_MODEL * D_MODEL;
  const size_t nkv = (size_t)KVH * HD * D_MODEL;
  cvt_f16_kernel<<<dim3((unsigned)(nx / 1024)),  blk256, 0, stream>>>(x,  xh,  (int)(nx / 4));
  cvt_f16_kernel<<<dim3((unsigned)(nqo / 1024)), blk256, 0, stream>>>(wq, wqh, (int)(nqo / 4));
  cvt_f16_kernel<<<dim3((unsigned)(nkv / 1024)), blk256, 0, stream>>>(wk, wkh, (int)(nkv / 4));
  cvt_f16_kernel<<<dim3((unsigned)(nkv / 1024)), blk256, 0, stream>>>(wv, wvh, (int)(nkv / 4));
  cvt_f16_kernel<<<dim3((unsigned)(nqo / 1024)), blk256, 0, stream>>>(wo, woh, (int)(nqo / 4));

  qkv_rope_kernel<<<dim3(B_SZ * L_SEQ / 16, 48), blk32, 0, stream>>>(
      xh, wqh, wkh, wvh, Q, K, Vt);
  attn_kernel<<<dim3(L_SEQ / 16, NH, B_SZ), blk32, 0, stream>>>(Q, K, Vt, O);
  oproj_kernel<<<dim3(B_SZ * L_SEQ / 16, D_MODEL / 64), blk32, 0, stream>>>(
      O, woh, out);
}
